// GRU_neighbor_89945205113501
// MI455X (gfx1250) — matrix operand order repaired and checked by reading, not compiled, not run on hardware
//
#include <hip/hip_runtime.h>
#include <hip/hip_bf16.h>
#include <math.h>

#define TT   12
#define BB   1024
#define DD   64
#define S1N  25
#define S2N  10
#define TOTS 276   // 1 + 25 + 250

typedef __attribute__((ext_vector_type(16))) _Float16 v16h;
typedef __attribute__((ext_vector_type(8)))  float    v8f;

// ---------------- WMMA fragment helpers (16x16x32 f16; A/B in contiguous k order) ----------------
// A-matrix 16x32 f16: lanes 0-15 row M=lane hold K 0..15; lanes 16-31 row M=lane-16 hold K 16..31
__device__ inline v16h load_a_frag(const _Float16* A, int ldA, int mt, int kt, int lane) {
    int row = mt * 16 + (lane & 15);
    int kb  = kt * 32 + ((lane >> 4) << 4);   // +16 for upper lane half
    const _Float16* p = A + row * ldA + kb;
    v16h a;
#pragma unroll
    for (int h = 0; h < 16; h++) a[h] = p[h];
    return a;
}
// B-matrix 32x16 f16 (row-major K x N source): lanes 0-15 col N=lane K=0..15; lanes 16-31 col N=lane-16 K=16..31
__device__ inline v16h load_b_frag(const _Float16* B, int ldB, int kt, int nt, int lane) {
    int col = nt * 16 + (lane & 15);
    int kb  = kt * 32 + ((lane >> 4) << 4);   // +16 for upper lane half
    const _Float16* p = B + kb * ldB + col;
    v16h b;
#pragma unroll
    for (int h = 0; h < 16; h++) b[h] = p[h * ldB];
    return b;
}
// C/D 16x16 f32: lanes 0-15 M=r, lanes 16-31 M=r+8, N=lane&15
__device__ inline void store_c_frag(float* C, int ldC, int mt, int nt, int lane, v8f c) {
    int row0 = mt * 16 + ((lane >> 4) << 3);
    int col  = nt * 16 + (lane & 15);
#pragma unroll
    for (int r = 0; r < 8; r++) C[(row0 + r) * ldC + col] = c[r];
}

__device__ inline float sigmoidf_(float x) { return 1.0f / (1.0f + expf(-x)); }
__device__ inline float leakyf_(float x)   { return x >= 0.0f ? x : 0.01f * x; }
__device__ inline float wave_sum(float v)  {
#pragma unroll
    for (int off = 16; off; off >>= 1) v += __shfl_xor(v, off);
    return v;
}

// ---------------- prep: folded attention vectors + f16 transposed weights ----------------
__global__ void prep_kernel(const float* __restrict__ V1_h0, const float* __restrict__ w1_h0,
                            const float* __restrict__ V1_att, const float* __restrict__ w1_h1,
                            const float* __restrict__ V1_agg, const float* __restrict__ W_hops,
                            const float* __restrict__ Wz, const float* __restrict__ Wr1,
                            const float* __restrict__ Wr2_1, const float* __restrict__ Wr2_2,
                            const float* __restrict__ Wh_1, const float* __restrict__ Wh_2,
                            float* uA, float* uB, float* vaV, float* vbV,
                            _Float16* wBaggT, _Float16* wWhT, _Float16* wVh0T,
                            _Float16* wWzrT, _Float16* wWhTT, _Float16* wWr22T, _Float16* wWh2T)
{
    int tid = blockIdx.x * blockDim.x + threadIdx.x;
    if (tid < 64) {
        float a = 0.f, b = 0.f, c = 0.f, d = 0.f;
        for (int l = 0; l < 64; l++) {
            a += w1_h1[l]      * V1_att[l * 64 + tid];
            b += w1_h1[64 + l] * V1_att[l * 64 + tid];
            c += w1_h0[l]      * V1_h0[l * 64 + tid];
            d += w1_h0[64 + l] * V1_h0[l * 64 + tid];
        }
        uA[tid] = a; uB[tid] = b; vaV[tid] = c; vbV[tid] = d;
    }
    if (tid < 4096) {               // [dd][l] = W[l][dd]
        int dd = tid >> 6, l = tid & 63;
        wBaggT[tid] = (_Float16)V1_agg[l * 64 + dd];
        wVh0T[tid]  = (_Float16)V1_h0[l * 64 + dd];
    }
    if (tid < 8192) {               // [m][l] = W_hops[l][m], m<128
        int m = tid >> 6, l = tid & 63;
        wWhT[tid] = (_Float16)W_hops[l * 128 + m];
    }
    if (tid < 36864) {              // WzrT[k][n], K=192, N=192 (rows: z | r1 | r2)
        int k = tid / 192, n = tid % 192;
        float v;
        if (n < 64)        v = Wz[n * 192 + k];
        else if (n < 128)  v = Wr1[(n - 64) * 192 + k];
        else { int r = n - 128; v = (k < 64) ? Wr2_1[r * 64 + k] : Wr2_2[r * 128 + (k - 64)]; }
        wWzrT[tid] = (_Float16)v;
    }
    if (tid < 12288) {              // WhT[k][n], K=192, N=64
        int k = tid >> 6, n = tid & 63;
        float v = (k < 64) ? Wh_1[n * 64 + k] : Wh_2[n * 128 + (k - 64)];
        wWhTT[tid] = (_Float16)v;
    }
    if (tid < 8192) {               // t=0 path, K=128, N=64
        int k = tid >> 6, n = tid & 63;
        wWr22T[tid] = (_Float16)Wr2_2[n * 128 + k];
        wWh2T[tid]  = (_Float16)Wh_2[n * 128 + k];
    }
}

// ---------------- aggregation: one WG per (t,b), all timesteps in parallel ----------------
__global__ __launch_bounds__(128) void agg_kernel(
    const float* __restrict__ features, const int* __restrict__ samples,
    const float* __restrict__ uA, const float* __restrict__ uB,
    const float* __restrict__ vaV, const float* __restrict__ vbV,
    const _Float16* __restrict__ wBaggT, const _Float16* __restrict__ wWhT,
    const _Float16* __restrict__ wVh0T,
    float* __restrict__ XNT, float* __restrict__ outBeta)
{
    const int b = blockIdx.x, t = blockIdx.y;
    const int tid = threadIdx.x, lane = tid & 31, wave = tid >> 5;

    __shared__ _Float16 sA1[32 * 64];
    __shared__ _Float16 sA2[32 * 64];
    __shared__ float    sC1[32 * 64];
    __shared__ float    sC2[32 * 64];
    __shared__ float    sEn0[32];
    __shared__ float    sB0[32];
    __shared__ float    sGn[32];
    __shared__ float    sBh[32];
    __shared__ float    sEt0;

    // zero pad rows 26..31
    for (int i = tid; i < 6 * 64; i += 128) { sA1[26 * 64 + i] = (_Float16)0.f; sA2[26 * 64 + i] = (_Float16)0.f; }

    const float2 ua = *(const float2*)(uA + 2 * lane);
    const float2 ub = *(const float2*)(uB + 2 * lane);
    const int* smp = samples + (size_t)b * TOTS;

    // ---- phase 1: gather + level-1 attention; build A matrices (f16) ----
    for (int s = wave; s <= 25; s += 4) {
        if (s < 25) {
            // issue all 11 row gathers up front (keep loads in flight)
            int i1 = smp[1 + s];
            const int* i2 = smp + 1 + S1N + s * S2N;
            int idx2v[10];
#pragma unroll
            for (int k = 0; k < 10; k++) idx2v[k] = i2[k];
            float2 f1v = *(const float2*)(features + ((size_t)i1 * TT + t) * DD + 2 * lane);
            float2 rf[10];
#pragma unroll
            for (int k = 0; k < 10; k++)
                rf[k] = *(const float2*)(features + ((size_t)idx2v[k] * TT + t) * DD + 2 * lane);

            float pt  = wave_sum(f1v.x * ua.x + f1v.y * ua.y);   // e_t[b,s]
            float pn0 = wave_sum(f1v.x * ub.x + f1v.y * ub.y);   // e_n0[b,s]
            float en[10];
#pragma unroll
            for (int k = 0; k < 10; k++)
                en[k] = wave_sum(rf[k].x * ub.x + rf[k].y * ub.y);

            float mx = -1e30f;
#pragma unroll
            for (int k = 0; k < 10; k++) { en[k] = leakyf_(pt + en[k]); mx = fmaxf(mx, en[k]); }
            float ssum = 0.f;
#pragma unroll
            for (int k = 0; k < 10; k++) { en[k] = expf(en[k] - mx); ssum += en[k]; }
            float inv = 1.f / ssum;
            float2 wf = {0.f, 0.f};
#pragma unroll
            for (int k = 0; k < 10; k++) {
                float bt = en[k] * inv;
                wf.x += bt * rf[k].x; wf.y += bt * rf[k].y;
                if (lane == k) outBeta[(((size_t)t * BB + b) * S1N + s) * 11 + 1 + k] = bt;
            }
            sA1[s * 64 + 2 * lane] = (_Float16)f1v.x; sA1[s * 64 + 2 * lane + 1] = (_Float16)f1v.y;
            sA2[s * 64 + 2 * lane] = (_Float16)wf.x;  sA2[s * 64 + 2 * lane + 1] = (_Float16)wf.y;
            if (lane == 0) sEn0[s] = pn0;
        } else {
            int it = smp[0];
            const float* rt = features + ((size_t)it * TT + t) * DD;
            float2 fv = *(const float2*)(rt + 2 * lane);
            float pt = wave_sum(fv.x * ua.x + fv.y * ua.y);      // e_t0
            sA1[25 * 64 + 2 * lane] = (_Float16)fv.x; sA1[25 * 64 + 2 * lane + 1] = (_Float16)fv.y;
            sA2[25 * 64 + 2 * lane] = (_Float16)0.f;  sA2[25 * 64 + 2 * lane + 1] = (_Float16)0.f;
            if (lane == 0) sEt0 = pt;
        }
    }
    __syncthreads();

    // ---- beta0 (not an output; feeds agg1) ----
    if (wave == 0) {
        float v = (lane < 25) ? leakyf_(sEt0 + sEn0[lane]) : -1e30f;
        float mx = v;
#pragma unroll
        for (int off = 16; off; off >>= 1) mx = fmaxf(mx, __shfl_xor(mx, off));
        float e = (lane < 25) ? expf(v - mx) : 0.f;
        float ss = wave_sum(e);
        sB0[lane] = e / ss;
    }
    __syncthreads();

    // ---- matmul 1: emb1 = A1 @ V1_agg^T -> sC1 ; agg2 = A2 @ V1_agg^T -> sC2 ----
    for (int j = wave; j < 16; j += 4) {
        const _Float16* Am = (j < 8) ? sA1 : sA2;
        float* Cm = (j < 8) ? sC1 : sC2;
        int tle = j & 7, mt = tle >> 2, nt = tle & 3;
        v8f c = {};
#pragma unroll
        for (int kt = 0; kt < 2; kt++) {
            v16h a = load_a_frag(Am, 64, mt, kt, lane);
            v16h bf = load_b_frag(wBaggT, 64, kt, nt, lane);
            c = __builtin_amdgcn_wmma_f32_16x16x32_f16(false, a, false, bf, (short)0, c, false, false);
        }
        store_c_frag(Cm, 64, mt, nt, lane, c);
    }
    __syncthreads();

    // ---- agg1 into row 25 of sC2; convert both to f16 A buffers ----
    if (tid < 64) {
        float acc = 0.f;
        for (int s = 0; s < 25; s++) acc += sB0[s] * sC1[s * 64 + tid];
        sC2[25 * 64 + tid] = acc;
    }
    __syncthreads();
    for (int i = tid; i < 26 * 64; i += 128) { sA1[i] = (_Float16)sC1[i]; sA2[i] = (_Float16)sC2[i]; }
    __syncthreads();

    // ---- matmul 2: hopPre = [emb1|agg2] @ W_hops^T (K=128) -> sC1 ----
    for (int j = wave; j < 8; j += 4) {
        int mt = j >> 2, nt = j & 3;
        v8f c = {};
#pragma unroll
        for (int kt = 0; kt < 2; kt++) {
            v16h a = load_a_frag(sA1, 64, mt, kt, lane);
            v16h bf = load_b_frag(wWhT, 64, kt, nt, lane);
            c = __builtin_amdgcn_wmma_f32_16x16x32_f16(false, a, false, bf, (short)0, c, false, false);
        }
#pragma unroll
        for (int kt = 0; kt < 2; kt++) {
            v16h a = load_a_frag(sA2, 64, mt, kt, lane);
            v16h bf = load_b_frag(wWhT + 64 * 64, 64, kt, nt, lane);
            c = __builtin_amdgcn_wmma_f32_16x16x32_f16(false, a, false, bf, (short)0, c, false, false);
        }
        store_c_frag(sC1, 64, mt, nt, lane, c);
    }
    __syncthreads();

    // ---- sigmoid -> hop (f32 in sC1, f16 in sA1) ----
    for (int i = tid; i < 26 * 64; i += 128) {
        float v = sigmoidf_(sC1[i]);
        sC1[i] = v; sA1[i] = (_Float16)v;
    }
    __syncthreads();

    // ---- g_n[s] = vb_v . hop1[s]; g_t = va_v . hop0 ----
    {
        const float2 vav = *(const float2*)(vaV + 2 * lane);
        const float2 vbv = *(const float2*)(vbV + 2 * lane);
        for (int s = wave; s <= 25; s += 4) {
            float2 h = *(const float2*)(sC1 + s * 64 + 2 * lane);
            float2 u = (s < 25) ? vbv : vav;
            float p = wave_sum(h.x * u.x + h.y * u.y);
            if (lane == 0) sGn[s] = p;
        }
    }

    // ---- matmul 3: emb_h1 = hop @ V1_h0^T -> sC2 ----
    for (int j = wave; j < 8; j += 4) {
        int mt = j >> 2, nt = j & 3;
        v8f c = {};
#pragma unroll
        for (int kt = 0; kt < 2; kt++) {
            v16h a = load_a_frag(sA1, 64, mt, kt, lane);
            v16h bf = load_b_frag(wVh0T, 64, kt, nt, lane);
            c = __builtin_amdgcn_wmma_f32_16x16x32_f16(false, a, false, bf, (short)0, c, false, false);
        }
        store_c_frag(sC2, 64, mt, nt, lane, c);
    }
    __syncthreads();

    // ---- beta_h + output Beta[...,0] ----
    if (wave == 0) {
        float gt = sGn[25];
        float v = (lane < 25) ? leakyf_(gt + sGn[lane]) : -1e30f;
        float mx = v;
#pragma unroll
        for (int off = 16; off; off >>= 1) mx = fmaxf(mx, __shfl_xor(mx, off));
        float e = (lane < 25) ? expf(v - mx) : 0.f;
        float ss = wave_sum(e);
        float bh = e / ss;
        sBh[lane] = bh;
        if (lane < 25) outBeta[(((size_t)t * BB + b) * S1N + lane) * 11 + 0] = bh;
    }
    __syncthreads();

    // ---- xnt = sum_s beta_h[s] * emb_h1[s,:] ----
    if (tid < 64) {
        float acc = 0.f;
        for (int s = 0; s < 25; s++) acc += sBh[s] * sC2[s * 64 + tid];
        XNT[((size_t)t * BB + b) * 64 + tid] = acc;
    }
}

// ---------------- GRU t=0 : 16 batch rows per WG ----------------
__global__ __launch_bounds__(128) void gru0_kernel(
    const float* __restrict__ features, const int* __restrict__ xIdx,
    const float* __restrict__ XNT,
    const _Float16* __restrict__ wWr22T, const _Float16* __restrict__ wWh2T,
    const float* __restrict__ br2, const float* __restrict__ bh,
    float* __restrict__ hOut)
{
    const int b0 = blockIdx.x * 16;
    const int tid = threadIdx.x, lane = tid & 31, wave = tid >> 5;
    __shared__ _Float16 sA[16 * 128];
    __shared__ float sC[16 * 64], sXn[16 * 64];

    for (int r = wave; r < 16; r += 4) {
        int b = b0 + r;
        const float* xrow = features + ((size_t)xIdx[b] * TT + 0) * DD;
        const float* nrow = XNT + (size_t)b * 64;  // t=0
        float2 xv = *(const float2*)(xrow + 2 * lane);
        float2 nv = *(const float2*)(nrow + 2 * lane);
        sXn[r * 64 + 2 * lane] = nv.x;  sXn[r * 64 + 2 * lane + 1] = nv.y;
        sA[r * 128 + 2 * lane] = (_Float16)xv.x;        sA[r * 128 + 2 * lane + 1] = (_Float16)xv.y;
        sA[r * 128 + 64 + 2 * lane] = (_Float16)nv.x;   sA[r * 128 + 64 + 2 * lane + 1] = (_Float16)nv.y;
    }
    __syncthreads();

    for (int nt = wave; nt < 4; nt += 4) {    // rt2pre = [xt|xnt] @ Wr2_2^T
        v8f c = {};
#pragma unroll
        for (int kt = 0; kt < 4; kt++) {
            v16h a = load_a_frag(sA, 128, 0, kt, lane);
            v16h bf = load_b_frag(wWr22T, 64, kt, nt, lane);
            c = __builtin_amdgcn_wmma_f32_16x16x32_f16(false, a, false, bf, (short)0, c, false, false);
        }
        store_c_frag(sC, 64, 0, nt, lane, c);
    }
    __syncthreads();

    for (int i = tid; i < 16 * 64; i += 128) {   // second half of A := rt2 * xnt
        int r = i >> 6, col = i & 63;
        float rt2 = sigmoidf_(sC[i] + br2[col]);
        sA[r * 128 + 64 + col] = (_Float16)(rt2 * sXn[i]);
    }
    __syncthreads();

    for (int nt = wave; nt < 4; nt += 4) {    // h0 = tanh([xt|rt2*xnt] @ Wh_2^T + bh)
        v8f c = {};
#pragma unroll
        for (int kt = 0; kt < 4; kt++) {
            v16h a = load_a_frag(sA, 128, 0, kt, lane);
            v16h bf = load_b_frag(wWh2T, 64, kt, nt, lane);
            c = __builtin_amdgcn_wmma_f32_16x16x32_f16(false, a, false, bf, (short)0, c, false, false);
        }
        store_c_frag(sC, 64, 0, nt, lane, c);
    }
    __syncthreads();

    for (int i = tid; i < 16 * 64; i += 128) {
        int r = i >> 6, col = i & 63;
        hOut[(size_t)(b0 + r) * TT * 64 + 0 * 64 + col] = tanhf(sC[i] + bh[col]);
    }
}

// ---------------- GRU step t>=1 : 16 batch rows per WG ----------------
__global__ __launch_bounds__(128) void gru_step_kernel(
    const float* __restrict__ features, const int* __restrict__ xIdx,
    const float* __restrict__ XNT,
    const _Float16* __restrict__ wWzrT, const _Float16* __restrict__ wWhTT,
    const float* __restrict__ bz, const float* __restrict__ br1,
    const float* __restrict__ br2, const float* __restrict__ bh,
    float* __restrict__ hOut, int t)
{
    const int b0 = blockIdx.x * 16;
    const int tid = threadIdx.x, lane = tid & 31, wave = tid >> 5;
    __shared__ _Float16 sA[16 * 192];
    __shared__ float sC[16 * 192];
    __shared__ float sH[16 * 64], sXn[16 * 64];

    for (int r = wave; r < 16; r += 4) {
        int b = b0 + r;
        const float* hrow = hOut + (size_t)b * TT * 64 + (size_t)(t - 1) * 64;
        const float* xrow = features + ((size_t)xIdx[b] * TT + t) * DD;
        const float* nrow = XNT + ((size_t)t * BB + b) * 64;
        float2 hv = *(const float2*)(hrow + 2 * lane);
        float2 xv = *(const float2*)(xrow + 2 * lane);
        float2 nv = *(const float2*)(nrow + 2 * lane);
        sH[r * 64 + 2 * lane] = hv.x;  sH[r * 64 + 2 * lane + 1] = hv.y;
        sXn[r * 64 + 2 * lane] = nv.x; sXn[r * 64 + 2 * lane + 1] = nv.y;
        sA[r * 192 + 2 * lane] = (_Float16)hv.x;        sA[r * 192 + 2 * lane + 1] = (_Float16)hv.y;
        sA[r * 192 + 64 + 2 * lane] = (_Float16)xv.x;   sA[r * 192 + 64 + 2 * lane + 1] = (_Float16)xv.y;
        sA[r * 192 + 128 + 2 * lane] = (_Float16)nv.x;  sA[r * 192 + 128 + 2 * lane + 1] = (_Float16)nv.y;
    }
    __syncthreads();

    // zr pre-act = [hp|xt|xnt] @ Wzr^T  (16x192, K=192)
    for (int nt = wave; nt < 12; nt += 4) {
        v8f c = {};
#pragma unroll
        for (int kt = 0; kt < 6; kt++) {
            v16h a = load_a_frag(sA, 192, 0, kt, lane);
            v16h bf = load_b_frag(wWzrT, 192, kt, nt, lane);
            c = __builtin_amdgcn_wmma_f32_16x16x32_f16(false, a, false, bf, (short)0, c, false, false);
        }
        store_c_frag(sC, 192, 0, nt, lane, c);
    }
    __syncthreads();

    // z kept f32 in sC[:, :64]; A := [rt1*hp | xt | rt2*xnt]
    for (int i = tid; i < 16 * 192; i += 128) {
        int r = i / 192, c_ = i % 192;
        float pre = sC[i];
        if (c_ < 64) {
            sC[i] = sigmoidf_(pre + bz[c_]);
        } else if (c_ < 128) {
            float rt1 = sigmoidf_(pre + br1[c_ - 64]);
            sA[r * 192 + (c_ - 64)] = (_Float16)(rt1 * sH[r * 64 + (c_ - 64)]);
        } else {
            float rt2 = sigmoidf_(pre + br2[c_ - 128]);
            sA[r * 192 + c_] = (_Float16)(rt2 * sXn[r * 64 + (c_ - 128)]);
        }
    }
    __syncthreads();

    // h~ pre-act into sC[:, 64:128]
    for (int nt = wave; nt < 4; nt += 4) {
        v8f c = {};
#pragma unroll
        for (int kt = 0; kt < 6; kt++) {
            v16h a = load_a_frag(sA, 192, 0, kt, lane);
            v16h bf = load_b_frag(wWhTT, 64, kt, nt, lane);
            c = __builtin_amdgcn_wmma_f32_16x16x32_f16(false, a, false, bf, (short)0, c, false, false);
        }
        store_c_frag(sC + 64, 192, 0, nt, lane, c);
    }
    __syncthreads();

    for (int i = tid; i < 16 * 64; i += 128) {
        int r = i >> 6, c_ = i & 63;
        float z = sC[r * 192 + c_];
        float htil = tanhf(sC[r * 192 + 64 + c_] + bh[c_]);
        float hp = sH[i];
        hOut[(size_t)(b0 + r) * TT * 64 + (size_t)t * 64 + c_] = (1.f - z) * hp + z * htil;
    }
}

extern "C" void kernel_launch(void* const* d_in, const int* in_sizes, int n_in,
                              void* d_out, int out_size, void* d_ws, size_t ws_size,
                              hipStream_t stream) {
    const int*   x        = (const int*)  d_in[0];
    const float* features = (const float*)d_in[1];
    const int*   samples  = (const int*)  d_in[2];
    const float* V1_h0    = (const float*)d_in[3];
    const float* w1_h0    = (const float*)d_in[4];
    const float* V1_att   = (const float*)d_in[5];
    const float* w1_h1    = (const float*)d_in[6];
    const float* V1_agg   = (const float*)d_in[7];
    const float* W_hops   = (const float*)d_in[8];
    const float* Wz       = (const float*)d_in[9];
    const float* Wr1      = (const float*)d_in[10];
    const float* Wr2_1    = (const float*)d_in[11];
    const float* Wr2_2    = (const float*)d_in[12];
    const float* Wh_1     = (const float*)d_in[13];
    const float* Wh_2     = (const float*)d_in[14];
    const float* bz       = (const float*)d_in[15];
    const float* br1      = (const float*)d_in[16];
    const float* br2      = (const float*)d_in[17];
    const float* bh       = (const float*)d_in[18];

    float* out = (float*)d_out;
    float* outBeta = out + (size_t)BB * TT * 64;     // Beta after h_out

    // workspace layout
    float* uA  = (float*)d_ws;
    float* uB  = uA + 64;
    float* vaV = uB + 64;
    float* vbV = vaV + 64;
    float* XNT = vbV + 64;                            // 12*1024*64 f32
    _Float16* wBaggT = (_Float16*)(XNT + (size_t)TT * BB * 64);
    _Float16* wWhT   = wBaggT + 64 * 64;
    _Float16* wVh0T  = wWhT + 128 * 64;
    _Float16* wWzrT  = wVh0T + 64 * 64;
    _Float16* wWhTT  = wWzrT + 192 * 192;
    _Float16* wWr22T = wWhTT + 192 * 64;
    _Float16* wWh2T  = wWr22T + 128 * 64;

    prep_kernel<<<144, 256, 0, stream>>>(V1_h0, w1_h0, V1_att, w1_h1, V1_agg, W_hops,
                                         Wz, Wr1, Wr2_1, Wr2_2, Wh_1, Wh_2,
                                         uA, uB, vaV, vbV,
                                         wBaggT, wWhT, wVh0T, wWzrT, wWhTT, wWr22T, wWh2T);

    dim3 gridA(BB, TT);
    agg_kernel<<<gridA, 128, 0, stream>>>(features, samples, uA, uB, vaV, vbV,
                                          wBaggT, wWhT, wVh0T, XNT, outBeta);

    gru0_kernel<<<BB / 16, 128, 0, stream>>>(features, x, XNT, wWr22T, wWh2T, br2, bh, out);

    for (int t = 1; t < TT; t++) {
        gru_step_kernel<<<BB / 16, 128, 0, stream>>>(features, x, XNT, wWzrT, wWhTT,
                                                     bz, br1, br2, bh, out, t);
    }
}
